// GCNN_46394236731922
// MI455X (gfx1250) — matrix operand order repaired and checked by reading, not compiled, not run on hardware
//
#include <hip/hip_runtime.h>

#define BB   4
#define NN   50000
#define EE   800000
#define CC   128
#define MM   (BB * NN)                    // 200000 rows; 200000/64 = 3125 blocks exactly

#define AGG_ELEMS  ((size_t)MM * CC)      // 25.6M floats
#define WSPLIT_ELEMS (8 * 4 * 32 * 16)    // 16384 bf16 per buffer (32 KB)

typedef __attribute__((ext_vector_type(16))) __bf16 v16bf;
typedef __attribute__((ext_vector_type(8)))  float  v8f;

__device__ __forceinline__ unsigned lds_off(const void* p) {
    // generic -> LDS(addrspace 3) -> 32-bit byte offset for async-to-LDS dest
    return (unsigned)(__SIZE_TYPE__)(const __attribute__((address_space(3))) void*)p;
}

// ---------------------------------------------------------------- zero agg
__global__ __launch_bounds__(256) void GCNN_zero_kernel(float4* __restrict__ agg, int n4) {
    int i = blockIdx.x * blockDim.x + threadIdx.x;
    if (i < n4) agg[i] = make_float4(0.f, 0.f, 0.f, 0.f);
}

// ---------------------------------------------------------------- pre-split W into bf16 hi/lo, WMMA B-register layout
// 1024 threads total: t -> (ntile 0..7, kchunk 0..3, lane 0..31), 16 bf16 each
__global__ __launch_bounds__(256) void GCNN_prepw_kernel(
    const float* __restrict__ Wm, __bf16* __restrict__ whi, __bf16* __restrict__ wlo)
{
    int t    = blockIdx.x * 256 + threadIdx.x;      // 0..1023
    int nt   = t >> 7;
    int kc   = (t >> 5) & 3;
    int lane = t & 31;
    int h    = lane >> 4;
    int l16  = lane & 15;
    int base = ((nt * 4 + kc) * 32 + lane) * 16;
    #pragma unroll
    for (int j = 0; j < 16; ++j) {
        // B packing (this code's k order, not AMD's table order): lane holds K = 16h + j for column n
        float w = Wm[(size_t)(kc * 32 + 16 * h + j) * CC + nt * 16 + l16];
        __bf16 hv = (__bf16)w;
        whi[base + j] = hv;
        wlo[base + j] = (__bf16)(w - (float)hv);
    }
}

// ---------------------------------------------------------------- SpMM (COO scatter-add)
__global__ __launch_bounds__(128) void GCNN_spmm_kernel(
    const float* __restrict__ x, const int* __restrict__ erow,
    const int* __restrict__ ecol, const float* __restrict__ eval,
    float* __restrict__ agg)
{
    const int b  = blockIdx.y;
    const int e0 = blockIdx.x * 64;
    const int c  = threadIdx.x;             // channel 0..127

    __shared__ int   s_row[64];
    __shared__ int   s_col[64];
    __shared__ float s_val[64];

    if (threadIdx.x < 64) {
        int e = e0 + threadIdx.x;
        if (e < EE) {
            s_row[threadIdx.x] = erow[e];
            s_col[threadIdx.x] = ecol[e];
            s_val[threadIdx.x] = eval[e];
        }
    }
    __syncthreads();

    const float* xb = x   + (size_t)b * NN * CC;
    float*       ab = agg + (size_t)b * NN * CC;

    const int cnt = (EE - e0) < 64 ? (EE - e0) : 64;
    for (int i = 0; i < cnt; ++i) {
        float msg = s_val[i] * xb[(size_t)s_col[i] * CC + c];   // coalesced, L2-resident gather
        atomicAdd(&ab[(size_t)s_row[i] * CC + c], msg);         // global_atomic_add_f32 (L2 RMW)
    }
}

// ---------------------------------------------------------------- GEMM + bias + ReLU via bf16x3 WMMA
// grid: M/64 blocks (3125), block: 256 threads = 8 waves.
// Block covers 64 rows; wave w: m-tile = w&3, n-tiles = [(w>>2)*4, +4).
// Each wave builds its A hi/lo fragments ONCE and reuses them for 48 WMMAs.
__global__ __launch_bounds__(256) void GCNN_gemm_kernel(
    const float* __restrict__ agg,
    const __bf16* __restrict__ whi, const __bf16* __restrict__ wlo,
    const float* __restrict__ bias, float* __restrict__ out)
{
    __shared__ float sA[64 * CC];           // 32 KB agg slab, row-major [64][128]

    const int tid  = threadIdx.x;
    const int row0 = blockIdx.x * 64;

    // Async-copy agg tile -> LDS (ASYNCcnt path): 256 threads x 128 B each
    {
        const float* src = agg + (size_t)row0 * CC + tid * 32;
        unsigned     dst = lds_off(&sA[tid * 32]);
        asm volatile(
            "global_load_async_to_lds_b128 %0, %1, off\n\t"
            "global_load_async_to_lds_b128 %0, %1, off offset:16\n\t"
            "global_load_async_to_lds_b128 %0, %1, off offset:32\n\t"
            "global_load_async_to_lds_b128 %0, %1, off offset:48\n\t"
            "global_load_async_to_lds_b128 %0, %1, off offset:64\n\t"
            "global_load_async_to_lds_b128 %0, %1, off offset:80\n\t"
            "global_load_async_to_lds_b128 %0, %1, off offset:96\n\t"
            "global_load_async_to_lds_b128 %0, %1, off offset:112"
            :: "v"(dst), "v"(src) : "memory");
        asm volatile("s_wait_asynccnt 0x0" ::: "memory");
    }
    __syncthreads();

    const int wave = tid >> 5;
    const int lane = tid & 31;
    const int h    = lane >> 4;
    const int l16  = lane & 15;
    const int mt   = wave & 3;              // m-tile within the 64-row slab
    const int ntb  = (wave >> 2) * 4;       // first n-tile of this wave's half

    const float* sArow = &sA[(mt * 16 + l16) * CC];

    // Build A hi/lo fragments once (bf16 split of this wave's 16 rows)
    v16bf Ahi[4], Alo[4];
    #pragma unroll
    for (int kc = 0; kc < 4; ++kc) {
        #pragma unroll
        for (int j = 0; j < 16; ++j) {
            // A layout (16-bit A 16x32): K = 16h + j within chunk (matches B packing)
            int ka = kc * 32 + 16 * h + j;
            float a = sArow[ka];
            __bf16 av = (__bf16)a;
            Ahi[kc][j] = av;
            Alo[kc][j] = (__bf16)(a - (float)av);
        }
    }

    #pragma unroll
    for (int t = 0; t < 4; ++t) {
        const int nt = ntb + t;
        const int n  = nt * 16 + l16;
        v8f acc = {};
        #pragma unroll
        for (int kc = 0; kc < 4; ++kc) {
            size_t widx = (size_t)((nt * 4 + kc) * 32 + lane) * 16;
            v16bf bhi = *(const v16bf*)(whi + widx);   // 32 B contiguous per lane
            v16bf blo = *(const v16bf*)(wlo + widx);
            // bf16x3 split product, f32 accumulate (drop lo*lo term)
            acc = __builtin_amdgcn_wmma_f32_16x16x32_bf16(false, Ahi[kc], false, bhi, (short)0, acc, false, false);
            acc = __builtin_amdgcn_wmma_f32_16x16x32_bf16(false, Ahi[kc], false, blo, (short)0, acc, false, false);
            acc = __builtin_amdgcn_wmma_f32_16x16x32_bf16(false, Alo[kc], false, bhi, (short)0, acc, false, false);
        }
        const float bv = bias[n];
        #pragma unroll
        for (int i = 0; i < 8; ++i) {       // C/D layout: VGPR i -> row m = i + 8h
            int   m = mt * 16 + i + 8 * h;
            float v = acc[i] + bv;
            __builtin_nontemporal_store(v > 0.f ? v : 0.f,
                                        &out[(size_t)(row0 + m) * CC + n]);
        }
    }
}

// ---------------------------------------------------------------- launch
extern "C" void kernel_launch(void* const* d_in, const int* in_sizes, int n_in,
                              void* d_out, int out_size, void* d_ws, size_t ws_size,
                              hipStream_t stream) {
    const float* x    = (const float*)d_in[0];
    const int*   erow = (const int*)  d_in[1];
    const int*   ecol = (const int*)  d_in[2];
    const float* eval = (const float*)d_in[3];
    const float* Wm   = (const float*)d_in[4];
    const float* bias = (const float*)d_in[5];
    float*       out  = (float*)d_out;

    float*  agg = (float*)d_ws;                         // 102.4 MB
    __bf16* whi = (__bf16*)((char*)d_ws + AGG_ELEMS * sizeof(float));
    __bf16* wlo = whi + WSPLIT_ELEMS;                   // +32 KB each

    const int n4 = (int)(AGG_ELEMS / 4);
    GCNN_zero_kernel<<<(n4 + 255) / 256, 256, 0, stream>>>((float4*)agg, n4);

    GCNN_prepw_kernel<<<4, 256, 0, stream>>>(Wm, whi, wlo);

    dim3 gspmm((EE + 63) / 64, BB);
    GCNN_spmm_kernel<<<gspmm, 128, 0, stream>>>(x, erow, ecol, eval, agg);

    GCNN_gemm_kernel<<<MM / 64, 256, 0, stream>>>(agg, whi, wlo, bias, out);
}
